// NonZeroFeatureExtractor_40338332844248
// MI455X (gfx1250) — matrix operand order repaired and checked by reading, not compiled, not run on hardware
//
#include <hip/hip_runtime.h>

typedef __attribute__((ext_vector_type(16))) _Float16 v16h;
typedef __attribute__((ext_vector_type(8)))  _Float16 v8h;
typedef __attribute__((ext_vector_type(8)))  float    v8f;

// ---------------------------------------------------------------------------
// Pack conv weights f32 [Cout, Cin, 3, 3] -> f16 [Cout, KPAD], zero padded.
// CHM=0 (row-major K):    k = cin*9 + pos         (pos = dy*3+dx)
// CHM=1 (channel-minor K): k = pos*Cin + cin  -> B-fragment halfs become
//                          contiguous channel runs in a channels-last LDS tile.
// ---------------------------------------------------------------------------
template<int Cout, int Cin, int KPAD, int CHM>
__global__ void pack_w_k(const float* __restrict__ w, _Float16* __restrict__ out) {
  int idx = blockIdx.x * blockDim.x + threadIdx.x;
  if (idx >= Cout * KPAD) return;
  int m = idx / KPAD, k = idx - m * KPAD;
  float v = 0.0f;
  if (CHM) {
    int pos = k / Cin, cin = k - pos * Cin;
    if (pos < 9) v = w[(m * Cin + cin) * 9 + pos];
  } else {
    if (k < Cin * 9) v = w[m * Cin * 9 + k];
  }
  out[idx] = (_Float16)v;
}

__global__ void f32_to_f16_k(const float* __restrict__ in, _Float16* __restrict__ out, int n) {
  int i = blockIdx.x * blockDim.x + threadIdx.x;
  if (i < n) out[i] = (_Float16)in[i];
}

// ---------------------------------------------------------------------------
// Implicit-GEMM 3x3 SAME conv (+bias+ReLU, optionally fused 2x2 maxpool) via
// v_wmma_f32_16x16x32_f16, im2col operand staged in LDS.
//
// Workgroup = 8 waves = MT (Cout/16) x NSEG (16-wide x segments) sharing one
// LDS halo tile. POOL=1: each wave computes 2 conv rows, pools across the row
// pair + lane pair (__shfl_xor) and writes the pooled map directly (4x less
// HBM traffic on the big early layers).
//
// Cin >= 16 (all powers of two here): channels-last tile [row][col][CinP],
// CinP = Cin+8 -> per-lane B fragment = 16 contiguous halfs = two 16B v8h
// loads (ds_load_b128), conflict-free banking, no divergent guards (padded K
// chunks redirect the fragment base to a zeroed LDS slot via one cndmask).
//
// Fragment layouts used here (wave32; contiguous K, not the ISA table order):
//   A 16x32 f16 : lane L -> M = L%16 ; half h -> K = (L/16)*16 + h
//   B 32x16 f16 : lane L -> N = L%16 ; half h -> K = (L/16)*16 + h
//   C/D 16x16 f32: vgpr r -> M = r + (L/16)*8 ; N = L%16
// ---------------------------------------------------------------------------
template<int Cin, int Cout, int H, int W, int KPAD, int NSEG, int POOL>
__global__ __launch_bounds__(256) void conv3x3_wmma_k(
    const _Float16* __restrict__ in, const _Float16* __restrict__ wt,
    const float* __restrict__ bias, _Float16* __restrict__ out)
{
  constexpr int MT = Cout / 16;
  static_assert(MT * NSEG == 8, "workgroup must be 8 waves");
  constexpr int  CHM  = (Cin >= 16);
  constexpr int  ROWS = POOL ? 4 : 3;
  constexpr int  RYN  = POOL ? 2 : 1;
  constexpr int  LDSW = NSEG * 16 + 2;
  constexpr int  XSEGS = W / (16 * NSEG);
  constexpr int  HO = POOL ? H / 2 : H;
  constexpr int  WO = POOL ? W / 2 : W;
  constexpr int  CinP = Cin + 8;                      // 16B-aligned, bank-safe
  constexpr int  ZOFF = CHM ? ROWS * LDSW * CinP : Cin * ROWS * LDSW;

  __shared__ _Float16 tile[ZOFF + 16];

  const int tid  = threadIdx.x;
  const int lane = tid & 31;
  const int g    = lane >> 4;
  const int lm   = lane & 15;
  const int wv   = tid >> 5;
  const int mt   = wv % MT;
  const int seg  = wv / MT;

  int blk = blockIdx.x;
  const int xg = blk % XSEGS; blk /= XSEGS;
  const int yo = blk % HO;
  const int b  = blk / HO;
  const int ybase = POOL ? yo * 2 : yo;
  const int xbase = xg * (NSEG * 16);
  const int m0 = mt * 16;

  // ---- zero slot for padded-K fragments ----
  if (tid < 16) tile[ZOFF + tid] = (_Float16)0.0f;

  // ---- cooperative halo fill (coalesced global reads along cols) ----
  for (int idx = tid; idx < Cin * ROWS * LDSW; idx += 256) {
    int col = idx % LDSW;
    int rr  = idx / LDSW;
    int dy  = rr % ROWS;
    int cin = rr / ROWS;
    int iy = ybase + dy - 1;
    int ix = xbase + col - 1;
    _Float16 v = (_Float16)0.0f;
    if (iy >= 0 && iy < H && ix >= 0 && ix < W)
      v = in[(((long)b * Cin + cin) * H + iy) * W + ix];
    if (CHM) tile[(dy * LDSW + col) * CinP + cin] = v;
    else     tile[(cin * ROWS + dy) * LDSW + col] = v;
  }
  __syncthreads();

  v8f acc[RYN] = {};

  #pragma unroll
  for (int kc = 0; kc < KPAD; kc += 32) {
    // A: weights, two contiguous 8-half runs -> global_load_b128 pairs
    v16h av;
    #pragma unroll
    for (int h = 0; h < 16; ++h) {
      int k = kc + g * 16 + h;
      av[h] = wt[(m0 + lm) * KPAD + k];
    }
    #pragma unroll
    for (int ry = 0; ry < RYN; ++ry) {
      v16h bv;
      if (CHM) {
        // 16 contiguous halfs: one pos, channels cin0..cin0+15
        int k0   = kc + g * 16;          // Cin is a power of two -> shifts
        int pos  = k0 / Cin;
        int cin0 = k0 - pos * Cin;
        int dyk  = pos / 3;
        int dx   = pos - dyk * 3;
        int base = (pos < 9)
                 ? ((ry + dyk) * LDSW + seg * 16 + lm + dx) * CinP + cin0
                 : ZOFF;                 // padded chunk -> zero slot
        v8h lo = *(const v8h*)(tile + base);
        v8h hi = *(const v8h*)(tile + base + 8);
        #pragma unroll
        for (int h = 0; h < 16; ++h) bv[h] = (h < 8) ? lo[h] : hi[h - 8];
      } else {
        // Cin==1: tiny layer, per-element with zero-slot select (no EXEC ops)
        #pragma unroll
        for (int h = 0; h < 16; ++h) {
          int k = kc + g * 16 + h;
          int cin = k / 9, r = k - cin * 9;
          int dyk = r / 3, dx = r - dyk * 3;
          int idx = (k < Cin * 9)
                  ? (cin * ROWS + (ry + dyk)) * LDSW + seg * 16 + lm + dx
                  : ZOFF;
          bv[h] = tile[idx];
        }
      }
      acc[ry] = __builtin_amdgcn_wmma_f32_16x16x32_f16(
          false, av, false, bv, (short)0, acc[ry], false, false);
    }
  }

  // ---- epilogue: bias + ReLU (+ fused 2x2 maxpool) ----
  if (POOL) {
    #pragma unroll
    for (int r = 0; r < 8; ++r) {
      int m = m0 + r + g * 8;
      float bz = bias[m];
      float v0 = fmaxf(acc[0][r] + bz, 0.0f);
      float v1 = fmaxf(acc[RYN - 1][r] + bz, 0.0f);
      float vm = fmaxf(v0, v1);                 // max over row pair
      float other = __shfl_xor(vm, 1, 32);      // max over x pair
      if ((lm & 1) == 0) {
        float p = fmaxf(vm, other);
        int xo = (xbase + seg * 16 + lm) >> 1;
        out[(((long)b * Cout + m) * HO + yo) * WO + xo] = (_Float16)p;
      }
    }
  } else {
    #pragma unroll
    for (int r = 0; r < 8; ++r) {
      int m = m0 + r + g * 8;
      float v = fmaxf(acc[0][r] + bias[m], 0.0f);
      out[(((long)b * Cout + m) * HO + yo) * WO + (xbase + seg * 16 + lm)] =
          (_Float16)v;
    }
  }
}

// ---------------------------------------------------------------------------
// Global mean over H*W -> f32, strided/offset write lands inside concat buffer
// ---------------------------------------------------------------------------
__global__ void mean_hw_k(const _Float16* __restrict__ in, float* __restrict__ out,
                          int C, int HW, int out_ld, int out_off) {
  __shared__ float sm[256];
  int bc = blockIdx.x;
  float s = 0.0f;
  long base = (long)bc * HW;
  for (int t = threadIdx.x; t < HW; t += 256) s += (float)in[base + t];
  sm[threadIdx.x] = s; __syncthreads();
  for (int off = 128; off > 0; off >>= 1) {
    if (threadIdx.x < (unsigned)off) sm[threadIdx.x] += sm[threadIdx.x + off];
    __syncthreads();
  }
  if (threadIdx.x == 0) {
    int b = bc / C, c = bc - b * C;
    out[b * out_ld + out_off + c] = sm[0] / (float)HW;
  }
}

// ---------------------------------------------------------------------------
// Mask global stats: area, centroid, bbox over 256x256 -> stats[B,6]
// ---------------------------------------------------------------------------
__global__ void stats_k(const float* __restrict__ mask, float* __restrict__ stats) {
  __shared__ float s_a[256], s_si[256], s_sj[256], s_mi[256], s_Mi[256], s_mj[256], s_Mj[256];
  int b = blockIdx.x;
  const float* m = mask + (long)b * 65536;
  float a = 0.f, si = 0.f, sj = 0.f;
  float mi = 1e9f, Mi = -1e9f, mj = 1e9f, Mj = -1e9f;
  for (int p = threadIdx.x; p < 65536; p += 256) {
    if (m[p] != 0.0f) {
      float i = (float)(p >> 8), j = (float)(p & 255);
      a += 1.0f; si += i; sj += j;
      mi = fminf(mi, i); Mi = fmaxf(Mi, i);
      mj = fminf(mj, j); Mj = fmaxf(Mj, j);
    }
  }
  int t = threadIdx.x;
  s_a[t]=a; s_si[t]=si; s_sj[t]=sj; s_mi[t]=mi; s_Mi[t]=Mi; s_mj[t]=mj; s_Mj[t]=Mj;
  __syncthreads();
  for (int off = 128; off > 0; off >>= 1) {
    if (t < off) {
      s_a[t]  += s_a[t+off];  s_si[t] += s_si[t+off]; s_sj[t] += s_sj[t+off];
      s_mi[t] = fminf(s_mi[t], s_mi[t+off]); s_Mi[t] = fmaxf(s_Mi[t], s_Mi[t+off]);
      s_mj[t] = fminf(s_mj[t], s_mj[t+off]); s_Mj[t] = fmaxf(s_Mj[t], s_Mj[t+off]);
    }
    __syncthreads();
  }
  if (t == 0) {
    float area = s_a[0];
    float* o = stats + b * 6;
    if (area > 0.0f) {
      float cy = s_si[0] / area, cx = s_sj[0] / area;
      float hh = s_Mi[0] - s_mi[0], ww = s_Mj[0] - s_mj[0];
      o[0] = cy / 256.f; o[1] = cx / 256.f; o[2] = hh / 256.f; o[3] = ww / 256.f;
      o[4] = area / 65536.f; o[5] = hh * ww / 65536.f;
    } else {
      o[0]=o[1]=o[2]=o[3]=o[4]=o[5]=0.0f;
    }
  }
}

// ---------------------------------------------------------------------------
// Tiny dense layers (<= 64x256x256 MACs -> negligible; plain VALU).
// ---------------------------------------------------------------------------
__global__ void linear_k(const float* __restrict__ in, int in_ld,
                         const float* __restrict__ w, const float* __restrict__ bias,
                         float* __restrict__ out, int out_ld, int out_off,
                         int IN, int OUT, int relu, int total) {
  int idx = blockIdx.x * blockDim.x + threadIdx.x;
  if (idx >= total) return;
  int b = idx / OUT, o = idx - b * OUT;
  float acc = bias[o];
  const float* ip = in + b * in_ld;
  const float* wp = w + o * IN;
  for (int i = 0; i < IN; ++i) acc += ip[i] * wp[i];
  if (relu) acc = fmaxf(acc, 0.0f);
  out[b * out_ld + out_off + o] = acc;
}

// ---------------------------------------------------------------------------
extern "C" void kernel_launch(void* const* d_in, const int* in_sizes, int n_in,
                              void* d_out, int out_size, void* d_ws, size_t ws_size,
                              hipStream_t stream) {
  const float* x    = (const float*)d_in[0];
  const float* mask = (const float*)d_in[1];
  const float* lw1  = (const float*)d_in[2];  const float* lb1 = (const float*)d_in[3];
  const float* lw2  = (const float*)d_in[4];  const float* lb2 = (const float*)d_in[5];
  const float* lw3  = (const float*)d_in[6];  const float* lb3 = (const float*)d_in[7];
  const float* sw1  = (const float*)d_in[8];  const float* sb1 = (const float*)d_in[9];
  const float* sw2  = (const float*)d_in[10]; const float* sb2 = (const float*)d_in[11];
  const float* sw3  = (const float*)d_in[12]; const float* sb3 = (const float*)d_in[13];
  const float* gw   = (const float*)d_in[14]; const float* gb  = (const float*)d_in[15];
  const float* fpw1 = (const float*)d_in[16]; const float* fpb1= (const float*)d_in[17];
  const float* fpw2 = (const float*)d_in[18]; const float* fpb2= (const float*)d_in[19];
  const float* ffw1 = (const float*)d_in[20]; const float* ffb1= (const float*)d_in[21];
  const float* ffw2 = (const float*)d_in[22]; const float* ffb2= (const float*)d_in[23];
  float* out = (float*)d_out;

  // ---- workspace (~110 MB) ----
  char* wsb = (char*)d_ws;
  _Float16* A  = (_Float16*)(wsb);                          // 67,108,864 B
  _Float16* Bp = (_Float16*)(wsb + 67108864);               // 33,554,432 B
  _Float16* Cc = (_Float16*)(wsb + 67108864 + 33554432);    //  8,388,608 B (f16 input)
  char* wreg = wsb + 67108864 + 33554432 + 8388608;
  _Float16* wl1 = (_Float16*)wreg;            // 32*32
  _Float16* wl2 = wl1 + 32*32;                // 64*288
  _Float16* wl3 = wl2 + 64*288;               // 128*576
  _Float16* ws1 = wl3 + 128*576;              // 16*32
  _Float16* ws2 = ws1 + 16*32;                // 32*160
  _Float16* ws3 = ws2 + 32*160;               // 64*288
  float* freg     = (float*)(ws3 + 64*288);
  float* stats    = freg;                 // 64*6
  float* combined = stats + 64*6;         // 64*80  = [conv_feats(64) | glob(16)]
  float* h1       = combined + 64*80;     // 64*128
  float* fused    = h1 + 64*128;          // 64*192 = [local_feats(128) | shape(64)]
  float* g1       = fused + 64*192;       // 64*256

  // ---- pack weights (K-order matches each conv's fragment path) ----
  pack_w_k<32,1,32,0>    <<<(32*32   + 255)/256, 256, 0, stream>>>(lw1, wl1);
  pack_w_k<64,32,288,1>  <<<(64*288  + 255)/256, 256, 0, stream>>>(lw2, wl2);
  pack_w_k<128,64,576,1> <<<(128*576 + 255)/256, 256, 0, stream>>>(lw3, wl3);
  pack_w_k<16,1,32,0>    <<<(16*32   + 255)/256, 256, 0, stream>>>(sw1, ws1);
  pack_w_k<32,16,160,1>  <<<(32*160  + 255)/256, 256, 0, stream>>>(sw2, ws2);
  pack_w_k<64,32,288,1>  <<<(64*288  + 255)/256, 256, 0, stream>>>(sw3, ws3);

  const int nx = 64*256*256;

  // ---- x branch: conv+relu+pool fused, conv3 + mean ----
  f32_to_f16_k<<<(nx+255)/256, 256, 0, stream>>>(x, Cc, nx);
  conv3x3_wmma_k<1,32,256,256,32,4,1>   <<<64*128*4, 256, 0, stream>>>(Cc, wl1, lb1, A);
  conv3x3_wmma_k<32,64,128,128,288,2,1> <<<64*64*4,  256, 0, stream>>>(A,  wl2, lb2, Bp);
  conv3x3_wmma_k<64,128,64,64,576,1,0>  <<<64*64*4,  256, 0, stream>>>(Bp, wl3, lb3, A);
  mean_hw_k<<<64*128, 256, 0, stream>>>(A, fused, 128, 4096, 192, 0);      // local_feats

  // ---- mask branch ----
  f32_to_f16_k<<<(nx+255)/256, 256, 0, stream>>>(mask, Cc, nx);
  conv3x3_wmma_k<1,16,256,256,32,8,1>   <<<64*128*2, 256, 0, stream>>>(Cc, ws1, sb1, A);
  conv3x3_wmma_k<16,32,128,128,160,4,1> <<<64*64*2,  256, 0, stream>>>(A,  ws2, sb2, Bp);
  conv3x3_wmma_k<32,64,64,64,288,2,0>   <<<64*64*2,  256, 0, stream>>>(Bp, ws3, sb3, A);
  mean_hw_k<<<64*64, 256, 0, stream>>>(A, combined, 64, 4096, 80, 0);      // conv_feats

  // ---- global stats + MLP head ----
  stats_k<<<64, 256, 0, stream>>>(mask, stats);
  linear_k<<<(64*16 +255)/256, 256, 0, stream>>>(stats,    6,   gw,   gb,   combined, 80,  64, 6,   16,  0, 64*16);
  linear_k<<<(64*128+255)/256, 256, 0, stream>>>(combined, 80,  fpw1, fpb1, h1,       128, 0,  80,  128, 1, 64*128);
  linear_k<<<(64*64 +255)/256, 256, 0, stream>>>(h1,       128, fpw2, fpb2, fused,    192, 128,128, 64,  0, 64*64);
  linear_k<<<(64*256+255)/256, 256, 0, stream>>>(fused,    192, ffw1, ffb1, g1,       256, 0,  192, 256, 1, 64*256);
  linear_k<<<(64*256+255)/256, 256, 0, stream>>>(g1,       256, ffw2, ffb2, out,      256, 0,  256, 256, 0, 64*256);
}
